// MultiheadAttentionWithRope_33681133535351
// MI455X (gfx1250) — compile-verified, matrix operand order checked by reading, not run on hardware
//
#include <hip/hip_runtime.h>
#include <hip/hip_bf16.h>

// ---- problem constants (match reference) ----
#define B_  2
#define S_  2048
#define D_  1024
#define H_  16
#define HD_ 64
#define M_  (B_ * S_)        // 4096 tokens
#define N_QKV (3 * D_)       // 3072

typedef __attribute__((ext_vector_type(16))) _Float16 v16h;
typedef __attribute__((ext_vector_type(8)))  float    v8f;
typedef __attribute__((ext_vector_type(4)))  unsigned int u32x4;
typedef int v4i __attribute__((vector_size(4 * sizeof(int))));   // builtin's pointee type

#if defined(__gfx1250__) && __has_builtin(__builtin_amdgcn_global_load_async_to_lds_b128)
#define HAVE_ASYNC_LDS 1
#else
#define HAVE_ASYNC_LDS 0
#endif

// 16B global -> LDS copy. Async (ASYNCcnt-tracked, no VGPR staging) when the
// gfx1250 builtin is available; plain load/store otherwise.
__device__ __forceinline__ void cp16_to_lds(const _Float16* g, _Float16* l) {
#if HAVE_ASYNC_LDS
    __builtin_amdgcn_global_load_async_to_lds_b128(
        (__attribute__((address_space(1))) v4i*)g,
        (__attribute__((address_space(3))) v4i*)l, 0, 0);
#else
    *(u32x4*)l = *(const u32x4*)g;
#endif
}

__device__ __forceinline__ void wait_async_lds() {
#if HAVE_ASYNC_LDS
#if __has_builtin(__builtin_amdgcn_s_wait_asynccnt)
    __builtin_amdgcn_s_wait_asynccnt(0);
#else
    asm volatile("s_wait_asynccnt 0x0" ::: "memory");
#endif
#endif
}

__device__ __forceinline__ v8f vzero8() {
    v8f z = {0.f, 0.f, 0.f, 0.f, 0.f, 0.f, 0.f, 0.f};
    return z;
}

__device__ __forceinline__ v8f wmma_f16(v16h a, v16h b, v8f c) {
    // D = A(16x32 f16) * B(32x16 f16) + C(16x16 f32)
    return __builtin_amdgcn_wmma_f32_16x16x32_f16(
        /*neg_a=*/false, a, /*neg_b=*/false, b,
        /*c_mod=*/(short)0, c, /*reuse_a=*/false, /*reuse_b=*/false);
}

// Load a 16x32 f16 tile in the CDNA5 WMMA A/B per-lane layout from a
// row-major matrix (row stride `stride` halfs). lane&15 selects row,
// lane>>4 selects the 8-half sub-chunk: halfs [col0+off, +8) and
// [col0+off+16, +8). Works for global or LDS-backed pointers.
__device__ __forceinline__ v16h load_tile(const _Float16* base,
                                          int stride, int row0, int col0) {
    int lane = threadIdx.x & 31;
    const _Float16* p = base + (size_t)(row0 + (lane & 15)) * stride
                        + col0 + ((lane >> 4) ? 8 : 0);
    union { v16h v; u32x4 u[2]; } t;
    t.u[0] = *(const u32x4*)(p);
    t.u[1] = *(const u32x4*)(p + 16);
    return t.v;
}

// -------------------- kernel 0: f32 -> f16 convert --------------------
__global__ void cvt_f32_to_f16(const float* __restrict__ in,
                               _Float16* __restrict__ out, int n) {
    for (int i = blockIdx.x * blockDim.x + threadIdx.x; i < n;
         i += gridDim.x * blockDim.x)
        out[i] = (_Float16)in[i];
}

// -------------------- kernel 1: LayerNorm -> f16 --------------------
__global__ void layernorm_f16(const float* __restrict__ x,
                              const float* __restrict__ gamma,
                              const float* __restrict__ beta,
                              _Float16* __restrict__ y) {
    const int D = D_;
    int row = blockIdx.x;
    const float* xr = x + (size_t)row * D;
    float s = 0.f, ss = 0.f;
    for (int i = threadIdx.x; i < D; i += blockDim.x) {
        float v = xr[i];
        s += v; ss += v * v;
    }
    for (int o = 16; o > 0; o >>= 1) {
        s  += __shfl_down(s,  o, 32);
        ss += __shfl_down(ss, o, 32);
    }
    __shared__ float red[2][8];
    int wv = threadIdx.x >> 5, ln = threadIdx.x & 31;
    if (ln == 0) { red[0][wv] = s; red[1][wv] = ss; }
    __syncthreads();
    if (threadIdx.x == 0) {
        float S = 0.f, SS = 0.f;
        for (int i = 0; i < (int)(blockDim.x >> 5); ++i) { S += red[0][i]; SS += red[1][i]; }
        red[0][0] = S; red[1][0] = SS;
    }
    __syncthreads();
    float mu  = red[0][0] / (float)D;
    float var = red[1][0] / (float)D - mu * mu;
    float rs  = rsqrtf(var + 1e-5f);
    for (int i = threadIdx.x; i < D; i += blockDim.x)
        y[(size_t)row * D + i] = (_Float16)((xr[i] - mu) * rs * gamma[i] + beta[i]);
}

// -------------------- kernels 2/5: WMMA GEMM, double-buffered async LDS --------------------
// A: [M,K] f16 row-major.  W: [N,K] f16 row-major.  bias: [N] f32.
// Block: 128 threads = 4 waves stacked in M; block tile 128M x 64N.
// W k-tiles (64n x 32k = 4KB) are async-copied into a double-buffered LDS
// region; next tile is in flight while 8 WMMAs/wave consume the current one.
template <typename OutT>
__global__ void gemm_wmma_bias(const _Float16* __restrict__ A,
                               const _Float16* __restrict__ W,
                               const float* __restrict__ bias,
                               OutT* __restrict__ C,
                               int M, int N, int K) {
    __shared__ __attribute__((aligned(16))) _Float16 Ws[2][64 * 32];
    int tid  = threadIdx.x;
    int wave = tid >> 5;
    int lane = tid & 31;
    int m0 = blockIdx.y * 128 + wave * 32;
    int n0 = blockIdx.x * 64;

    auto stage = [&](int buf, int k) {
        #pragma unroll
        for (int c = tid; c < 256; c += 128) {        // 256 x 16B chunks
            int row = c >> 2, off = (c & 3) * 8;
            cp16_to_lds(W + (size_t)(n0 + row) * K + k + off,
                        &Ws[buf][row * 32 + off]);
        }
    };

    v8f acc[2][4];
    #pragma unroll
    for (int s = 0; s < 2; ++s)
        #pragma unroll
        for (int t = 0; t < 4; ++t) acc[s][t] = vzero8();

    stage(0, 0);
    wait_async_lds();
    __syncthreads();

    for (int k = 0; k < K; k += 32) {
        int buf = (k >> 5) & 1;
        if (k + 32 < K) {
            stage(buf ^ 1, k + 32);                   // async prefetch next tile
            __builtin_prefetch(A + (size_t)(m0 + (lane & 15)) * K + k + 32, 0, 1);
        }
        v16h a0 = load_tile(A, K, m0,      k);
        v16h a1 = load_tile(A, K, m0 + 16, k);
        #pragma unroll
        for (int t = 0; t < 4; ++t) {
            v16h b = load_tile(&Ws[buf][0], 32, t * 16, 0);   // ds_load_b128 x2
            acc[0][t] = wmma_f16(a0, b, acc[0][t]);
            acc[1][t] = wmma_f16(a1, b, acc[1][t]);
        }
        wait_async_lds();
        __syncthreads();
    }

    int hs = lane >> 4;
    #pragma unroll
    for (int s = 0; s < 2; ++s)
        #pragma unroll
        for (int t = 0; t < 4; ++t) {
            int n = n0 + t * 16 + (lane & 15);
            float bv = bias[n];
            #pragma unroll
            for (int r = 0; r < 8; ++r) {
                int m = m0 + s * 16 + r + hs * 8;
                C[(size_t)m * N + n] = (OutT)(acc[s][t][r] + bv);
            }
        }
}

// -------------------- kernel 3: RoPE + scatter to Q, K, V^T --------------------
// qkv: [M, 3072] f16 (bias folded in by GEMM).
// q_r gets the 1/sqrt(HD) attention scale folded in here.
__global__ void rope_scatter(const _Float16* __restrict__ qkv,
                             _Float16* __restrict__ q_r,
                             _Float16* __restrict__ k_r,
                             _Float16* __restrict__ v_t) {
    int u = blockIdx.x * 4 + (threadIdx.x >> 6);   // (token, head) unit
    int d = threadIdx.x & 63;
    int h = u & 15;
    int m = u >> 4;                 // global token = b*S + s
    int s = m & (S_ - 1);
    int b = m >> 11;                // S_ = 2048
    const _Float16* row = qkv + (size_t)m * N_QKV + h * (3 * HD_);

    float qv  = (float)row[d];
    float qv2 = (float)row[d ^ 32];
    float kv  = (float)row[HD_ + d];
    float kv2 = (float)row[HD_ + (d ^ 32)];
    float vv  = (float)row[2 * HD_ + d];

    // inv_freq[i] = 10000^(-2i/64); log(10000)/32 = 0.28782313662425574
    float freq = __expf(-(float)(d & 31) * 0.28782313662425574f);
    float ang  = (float)s * freq;
    float c = cosf(ang), sn = sinf(ang);
    float rq = (d < 32) ? -qv2 : qv2;
    float rk = (d < 32) ? -kv2 : kv2;
    float qo = (qv * c + rq * sn) * 0.125f;        // fold 1/sqrt(64)
    float ko = kv * c + rk * sn;

    int bh = b * H_ + h;
    q_r[((size_t)bh * S_ + s) * HD_ + d] = (_Float16)qo;
    k_r[((size_t)bh * S_ + s) * HD_ + d] = (_Float16)ko;
    v_t[((size_t)bh * HD_ + d) * S_ + s] = (_Float16)vv;
}

// -------------------- kernel 4: flash attention, double-buffered async K/V --------------------
// One block owns a 128-row q tile of one (b,h); 8 waves each take 16 q rows,
// sharing one j-loop so K (32x64) and V^T (64x32) tiles are staged to LDS
// once per j-block and consumed by all waves. Tiles for j-block jb+1 are
// async-copied while WMMAs run on block jb (one wait+barrier per iteration).
// Scores are computed transposed (S^T = K*Q^T) so the f32 C layout (lane = q)
// matches the f16 A/B per-lane layout -> softmaxed P feeds the V^T*P WMMA
// with no cross-lane transpose; only a lane-xor-16 shuffle merges row halves.
__global__ void flash_attn(const _Float16* __restrict__ q_r,
                           const _Float16* __restrict__ k_r,
                           const _Float16* __restrict__ v_t,
                           _Float16* __restrict__ ctx) {
    __shared__ __attribute__((aligned(16))) _Float16 Ks[2][32 * 64];
    __shared__ __attribute__((aligned(16))) _Float16 Vs[2][64 * 32];
    int tid  = threadIdx.x;
    int wave = tid >> 5;
    int lane = tid & 31;
    int bh = blockIdx.y;                       // 0..31
    int q0 = blockIdx.x * 128;                 // block q tile
    int qw0 = q0 + wave * 16;                  // wave q tile

    const _Float16* qb = q_r + (size_t)bh * S_ * HD_;
    const _Float16* kb = k_r + (size_t)bh * S_ * HD_;
    const _Float16* vb = v_t + (size_t)bh * HD_ * S_;

    auto stage = [&](int buf, int j0) {
        #pragma unroll
        for (int c = tid; c < 512; c += 256) {        // 512 x 16B chunks
            if (c < 256) {           // K rows: 32 rows x 8 chunks
                int row = c >> 3, off = (c & 7) * 8;
                cp16_to_lds(kb + (size_t)(j0 + row) * HD_ + off,
                            &Ks[buf][row * 64 + off]);
            } else {                 // V^T rows: 64 rows x 4 chunks
                int cc = c - 256;
                int row = cc >> 2, off = (cc & 3) * 8;
                cp16_to_lds(vb + (size_t)row * S_ + j0 + off,
                            &Vs[buf][row * 32 + off]);
            }
        }
    };

    v16h Qb0 = load_tile(qb, HD_, qw0, 0);
    v16h Qb1 = load_tile(qb, HD_, qw0, 32);

    v8f O[4];
    #pragma unroll
    for (int t = 0; t < 4; ++t) O[t] = vzero8();
    float m_run = -1e30f, l_run = 0.f;

    int qg    = qw0 + (lane & 15);             // this lane's q row
    int joff  = (lane >> 4) * 8;
    int nfull = qw0 >> 5;                      // unmasked j-blocks for this wave
    int jblocks = (q0 >> 5) + 4;               // shared loop: j up to q0+127

    stage(0, 0);
    wait_async_lds();
    __syncthreads();

    for (int jb = 0; jb < jblocks; ++jb) {
        int j0 = jb * 32;
        int buf = jb & 1;
        if (jb + 1 < jblocks)
            stage(buf ^ 1, j0 + 32);           // async prefetch next K/V tiles

        const _Float16* ks = &Ks[buf][0];
        const _Float16* vs = &Vs[buf][0];

        v8f s0 = vzero8(), s1 = vzero8();
        s0 = wmma_f16(load_tile(ks, 64, 0,  0),  Qb0, s0);
        s0 = wmma_f16(load_tile(ks, 64, 0,  32), Qb1, s0);
        s1 = wmma_f16(load_tile(ks, 64, 16, 0),  Qb0, s1);
        s1 = wmma_f16(load_tile(ks, 64, 16, 32), Qb1, s1);

        float p0[8], p1[8];
        float mloc = -1e30f;
        if (jb < nfull) {            // fully-causal block: no mask VALU at all
            #pragma unroll
            for (int r = 0; r < 8; ++r) {
                p0[r] = s0[r]; p1[r] = s1[r];
                mloc = fmaxf(mloc, fmaxf(p0[r], p1[r]));
            }
        } else {                     // diagonal / beyond: apply causal mask
            #pragma unroll
            for (int r = 0; r < 8; ++r) {
                int ja = j0 + joff + r;
                int jc = ja + 16;
                float sa = (ja > qg) ? -1e30f : s0[r];
                float sb = (jc > qg) ? -1e30f : s1[r];
                p0[r] = sa; p1[r] = sb;
                mloc = fmaxf(mloc, fmaxf(sa, sb));
            }
        }
        mloc = fmaxf(mloc, __shfl_xor(mloc, 16, 32));  // merge q-row halves
        float m_new = fmaxf(m_run, mloc);
        float alpha = __expf(m_run - m_new);

        float lsum = 0.f;
        v16h P;
        #pragma unroll
        for (int r = 0; r < 8; ++r) {
            float e0 = __expf(p0[r] - m_new);
            float e1 = __expf(p1[r] - m_new);
            lsum += e0 + e1;
            P[r]     = (_Float16)e0;
            P[r + 8] = (_Float16)e1;
        }
        lsum += __shfl_xor(lsum, 16, 32);
        l_run = l_run * alpha + lsum;
        m_run = m_new;

        #pragma unroll
        for (int t = 0; t < 4; ++t) {
            O[t] = O[t] * alpha;                       // lane = q, exact rescale
            O[t] = wmma_f16(load_tile(vs, 32, t * 16, 0), P, O[t]);
        }

        wait_async_lds();                              // next tiles landed
        __syncthreads();
    }

    float inv_l = 1.0f / l_run;
    int b = bh >> 4, h = bh & 15;
    int q = qw0 + (lane & 15);
    _Float16* crow = ctx + ((size_t)(b * S_ + q) * D_) + h * HD_;
    int dbase = (lane >> 4) * 8;
    #pragma unroll
    for (int t = 0; t < 4; ++t) {
        union { u32x4 u; _Float16 hv[8]; } st;
        #pragma unroll
        for (int r = 0; r < 8; ++r) st.hv[r] = (_Float16)(O[t][r] * inv_l);
        *(u32x4*)(crow + t * 16 + dbase) = st.u;
    }
}

// -------------------- launch --------------------
extern "C" void kernel_launch(void* const* d_in, const int* in_sizes, int n_in,
                              void* d_out, int out_size, void* d_ws, size_t ws_size,
                              hipStream_t stream) {
    const float* hidden   = (const float*)d_in[0];
    const float* ln_gamma = (const float*)d_in[1];
    const float* ln_beta  = (const float*)d_in[2];
    const float* w_qkv    = (const float*)d_in[3];
    const float* b_qkv    = (const float*)d_in[4];
    const float* w_proj   = (const float*)d_in[5];
    const float* b_proj   = (const float*)d_in[6];
    float* out = (float*)d_out;

    char* ws = (char*)d_ws;
    size_t off = 0;
    auto alloc = [&](size_t bytes) { char* p = ws + off; off += (bytes + 255) & ~(size_t)255; return p; };
    _Float16* xh     = (_Float16*)alloc((size_t)M_ * D_ * 2);        // LN output
    _Float16* wqkvh  = (_Float16*)alloc((size_t)N_QKV * D_ * 2);     // qkv weight f16
    _Float16* wprojh = (_Float16*)alloc((size_t)D_ * D_ * 2);        // proj weight f16
    _Float16* qkvh   = (_Float16*)alloc((size_t)M_ * N_QKV * 2);     // qkv activations
    _Float16* q_r    = (_Float16*)alloc((size_t)B_ * H_ * S_ * HD_ * 2);
    _Float16* k_r    = (_Float16*)alloc((size_t)B_ * H_ * S_ * HD_ * 2);
    _Float16* v_t    = (_Float16*)alloc((size_t)B_ * H_ * HD_ * S_ * 2);
    _Float16* ctxh   = (_Float16*)alloc((size_t)M_ * D_ * 2);
    (void)ws_size; (void)in_sizes; (void)n_in; (void)out_size;

    // 0) weights -> f16
    cvt_f32_to_f16<<<2048, 256, 0, stream>>>(w_qkv,  wqkvh,  N_QKV * D_);
    cvt_f32_to_f16<<<1024, 256, 0, stream>>>(w_proj, wprojh, D_ * D_);

    // 1) LayerNorm -> f16
    layernorm_f16<<<M_, 256, 0, stream>>>(hidden, ln_gamma, ln_beta, xh);

    // 2) QKV GEMM (+bias), f16 out:  [4096,1024] x [3072,1024]^T
    gemm_wmma_bias<_Float16><<<dim3(N_QKV / 64, M_ / 128), 128, 0, stream>>>(
        xh, wqkvh, b_qkv, qkvh, M_, N_QKV, D_);

    // 3) RoPE + scatter to Q (pre-scaled), K, V^T
    rope_scatter<<<(B_ * S_ * H_) / 4, 256, 0, stream>>>(qkvh, q_r, k_r, v_t);

    // 4) flash attention -> ctx [B,S,D] f16
    flash_attn<<<dim3(S_ / 128, B_ * H_), 256, 0, stream>>>(q_r, k_r, v_t, ctxh);

    // 5) output projection (+bias), f32 out
    gemm_wmma_bias<float><<<dim3(D_ / 64, M_ / 128), 128, 0, stream>>>(
        ctxh, wprojh, b_proj, out, M_, D_, D_);
}
